// Attend_15229954031746
// MI455X (gfx1250) — matrix operand order repaired and checked by reading, not compiled, not run on hardware
//
#include <hip/hip_runtime.h>

typedef __attribute__((ext_vector_type(16))) _Float16 v16h;
typedef __attribute__((ext_vector_type(8)))  float    v8f;
typedef __attribute__((ext_vector_type(4)))  float    v4f;

#define WAVES 8
constexpr int Bc = 2, Hc = 8, Nc = 2048, Dc = 64, WIN = 256;

__global__ __launch_bounds__(32 * WAVES)
void attn_gc_kernel(const float* __restrict__ q,
                    const float* __restrict__ k,
                    const float* __restrict__ v,
                    const unsigned char* __restrict__ mask,
                    const int* __restrict__ gsz,
                    float* __restrict__ out)
{
    // per-wave raw-score staging tile, row-major [16 rows][32 keys] (16 KB total)
    __shared__ float ldsS[WAVES][16][32];

    const int lane = threadIdx.x & 31;
    const int wv   = threadIdx.x >> 5;
    const int tile = blockIdx.x * WAVES + wv;
    if (tile >= Bc * Hc * (Nc / 16)) return;   // wave-uniform exit: EXEC all-1s for WMMA

    const int qt = tile % (Nc / 16);
    const int h  = (tile / (Nc / 16)) % Hc;
    const int b  = tile / ((Nc / 16) * Hc);
    const int q0 = qt * 16;
    const int G  = gsz[0];
    const float scale = 0.125f;                // 1/sqrt(64)

    const float* __restrict__ Qp = q + (size_t)(b * Hc + h) * Nc * Dc;
    const float* __restrict__ Kp = k + (size_t)(b * Hc + h) * Nc * Dc;
    const float* __restrict__ Vp = v + (size_t)(b * Hc + h) * Nc * Dc;
    const unsigned char* __restrict__ Mp = mask + (size_t)b * Nc;

    const int m16 = lane & 15;   // A: row index / B,C,D: column-within-16
    const int hh  = lane >> 4;   // lane half

    // ---- Q tile -> f16 A layout via b128 loads, pre-scaled ----
    // A slot e (0..15) of slab: kk = (e<8 ? hh*8+e : 16+hh*8+(e-8)); contiguous runs of 8.
    v16h aq0, aq1;
    {
        const v4f* Q4 = (const v4f*)(Qp + (size_t)(q0 + m16) * Dc);
        v4f c0 = Q4[2*hh + 0],  c1 = Q4[2*hh + 1];    // d = hh*8 .. +7
        v4f c2 = Q4[2*hh + 4],  c3 = Q4[2*hh + 5];    // d = 16+hh*8 .. +7
        v4f c4 = Q4[2*hh + 8],  c5 = Q4[2*hh + 9];    // d = 32+hh*8 .. +7
        v4f c6 = Q4[2*hh + 12], c7 = Q4[2*hh + 13];   // d = 48+hh*8 .. +7
#pragma unroll
        for (int i = 0; i < 4; ++i) {
            aq0[i]      = (_Float16)(c0[i] * scale);
            aq0[4 + i]  = (_Float16)(c1[i] * scale);
            aq0[8 + i]  = (_Float16)(c2[i] * scale);
            aq0[12 + i] = (_Float16)(c3[i] * scale);
            aq1[i]      = (_Float16)(c4[i] * scale);
            aq1[4 + i]  = (_Float16)(c5[i] * scale);
            aq1[8 + i]  = (_Float16)(c6[i] * scale);
            aq1[12 + i] = (_Float16)(c7[i] * scale);
        }
    }

    // A-domain row owned by this lane (duplicated across the two halves)
    const int qrow = q0 + m16;
    const int qge  = (qrow / G + 1) * G;   // allowed: key < qge
    const int qlo  = qrow - WIN;           // allowed: key >= qlo

    v8f acc[4] = {v8f{}, v8f{}, v8f{}, v8f{}};
    float mrun = -1e30f;                   // finite sentinel -> branchless exp guards
    float lrun = 0.f;

    int kb_lo = q0 - WIN; if (kb_lo < 0) kb_lo = 0; kb_lo &= ~31;
    int kmax  = ((q0 + 15) / G) * G + G - 1; if (kmax > Nc - 1) kmax = Nc - 1;
    // invariant: kb is 32-aligned and kb <= kmax <= Nc-1  =>  kb+31 <= Nc-1 (no OOB)

    for (int kb = kb_lo; kb <= kmax; kb += 32) {
        if (kb + 32 <= kmax) {             // wave-uniform; prefetch next key block
            int pr = kb + 32 + lane; if (pr > Nc - 1) pr = Nc - 1;
            __builtin_prefetch(Kp + (size_t)pr * Dc, 0, 1);
            __builtin_prefetch(Vp + (size_t)pr * Dc, 0, 1);
        }

        // ---- S = (Q*scale) K^T : two 16-key sub-tiles, b128 K loads ----
        v8f s[2];
#pragma unroll
        for (int t = 0; t < 2; ++t) {
            const v4f* K4 = (const v4f*)(Kp + (size_t)(kb + 16 * t + m16) * Dc);
            v4f c0 = K4[2*hh + 0], c1 = K4[2*hh + 1], c2 = K4[2*hh + 4], c3 = K4[2*hh + 5];
            v4f c4 = K4[2*hh + 8], c5 = K4[2*hh + 9], c6 = K4[2*hh + 12], c7 = K4[2*hh + 13];
            v16h bk0, bk1;                 // B slot e: d = (e<8 ? hh*8+e : 16+hh*8+(e-8)) (+32 for slab 1)
#pragma unroll
            for (int i = 0; i < 4; ++i) {
                bk0[i]      = (_Float16)c0[i];
                bk0[4 + i]  = (_Float16)c1[i];
                bk0[8 + i]  = (_Float16)c2[i];
                bk0[12 + i] = (_Float16)c3[i];
                bk1[i]      = (_Float16)c4[i];
                bk1[4 + i]  = (_Float16)c5[i];
                bk1[8 + i]  = (_Float16)c6[i];
                bk1[12 + i] = (_Float16)c7[i];
            }
            v8f c = {};
            c = __builtin_amdgcn_wmma_f32_16x16x32_f16(false, aq0, false, bk0,
                                                       (short)0, c, false, false);
            c = __builtin_amdgcn_wmma_f32_16x16x32_f16(false, aq1, false, bk1,
                                                       (short)0, c, false, false);
            s[t] = c;
        }

        // ---- stage raw scores: C/D layout -> LDS row-major ----
#pragma unroll
        for (int t = 0; t < 2; ++t)
#pragma unroll
            for (int j = 0; j < 8; ++j)
                ldsS[wv][j + 8 * hh][16 * t + m16] = s[t][j];
        asm volatile("s_wait_dscnt 0" ::: "memory");   // intra-wave LDS exchange

        // ---- A-domain: lane sees 16 of row m16's 32 scores (partner = lane^16) ----
        const float* Srow = &ldsS[wv][m16][0];
        v4f r0 = *(const v4f*)(Srow + 8 * hh);
        v4f r1 = *(const v4f*)(Srow + 8 * hh + 4);
        v4f r2 = *(const v4f*)(Srow + 16 + 8 * hh);
        v4f r3 = *(const v4f*)(Srow + 16 + 8 * hh + 4);

        // padding-mask bytes for this lane's 16 slots (coalesced u32 loads)
        unsigned mw0 = *(const unsigned*)(Mp + kb + 8 * hh);
        unsigned mw1 = *(const unsigned*)(Mp + kb + 8 * hh + 4);
        unsigned mw2 = *(const unsigned*)(Mp + kb + 16 + 8 * hh);
        unsigned mw3 = *(const unsigned*)(Mp + kb + 16 + 8 * hh + 4);

        float sv[16];
#pragma unroll
        for (int i = 0; i < 16; ++i) {
            int key = kb + ((i < 8) ? (8 * hh + i) : (16 + 8 * hh + (i - 8)));
            unsigned mw = (i < 4) ? mw0 : (i < 8) ? mw1 : (i < 12) ? mw2 : mw3;
            float x = (i < 4) ? r0[i & 3] : (i < 8) ? r1[i & 3]
                    : (i < 12) ? r2[i & 3] : r3[i & 3];
            bool ok = (key < qge) && (key >= qlo) && (((mw >> (8 * (i & 3))) & 0xFF) != 0);
            sv[i] = ok ? x : -INFINITY;
        }

        // row max: local over 16 slots + one xor-16 with partner lane
        float mx = sv[0];
#pragma unroll
        for (int i = 1; i < 16; ++i) mx = fmaxf(mx, sv[i]);
        mx = fmaxf(mx, __shfl_xor(mx, 16, 32));
        float mnew = fmaxf(mrun, mx);
        float corr = __expf(mrun - mnew);   // safe: mrun,mnew >= -1e30

        // p = exp(s - m): lands directly in A layout for the PV matmul
        v16h ap;
        float ps = 0.f;
#pragma unroll
        for (int i = 0; i < 16; ++i) {
            float p = __expf(sv[i] - mnew); // -inf -> 0
            ps += p;
            ap[i] = (_Float16)p;
        }
        ps += __shfl_xor(ps, 16, 32);
        lrun = lrun * corr + ps;
        mrun = mnew;

        // broadcast corr (A-domain row r lives in lane r) to C/D-domain rows j+8*hh
#pragma unroll
        for (int j = 0; j < 8; ++j) {
            float cj = __shfl(corr, j + 8 * hh, 32);
#pragma unroll
            for (int tt = 0; tt < 4; ++tt) acc[tt][j] *= cj;
        }

        // ---- O += P V : 4 output-column tiles ----
        const float* Vb = Vp + (size_t)(kb + 8 * hh) * Dc + m16;  // B slot e: row kb+8*hh+(e<8 ? e : e+8)
#pragma unroll
        for (int tt = 0; tt < 4; ++tt) {
            v16h bv;
#pragma unroll
            for (int e = 0; e < 16; ++e)
                bv[e] = (_Float16)Vb[(size_t)(e < 8 ? e : e + 8) * Dc + 16 * tt];
            acc[tt] = __builtin_amdgcn_wmma_f32_16x16x32_f16(false, ap, false, bv,
                                                             (short)0, acc[tt], false, false);
        }
    }

    // ---- finalize: broadcast 1/l to C/D-domain rows, store fp32 ----
    float* __restrict__ Op = out + (size_t)(b * Hc + h) * Nc * Dc;
#pragma unroll
    for (int j = 0; j < 8; ++j) {
        float lj = __shfl(lrun, j + 8 * hh, 32);
        float rl = lj > 0.f ? 1.f / lj : 0.f;
#pragma unroll
        for (int tt = 0; tt < 4; ++tt)
            Op[(size_t)(q0 + j + 8 * hh) * Dc + 16 * tt + m16] = acc[tt][j] * rl;
    }
}

extern "C" void kernel_launch(void* const* d_in, const int* in_sizes, int n_in,
                              void* d_out, int out_size, void* d_ws, size_t ws_size,
                              hipStream_t stream) {
    (void)in_sizes; (void)n_in; (void)d_ws; (void)ws_size; (void)out_size;
    const float*         qp = (const float*)d_in[0];
    const float*         kp = (const float*)d_in[1];
    const float*         vp = (const float*)d_in[2];
    const unsigned char* mp = (const unsigned char*)d_in[3];  // jnp bool: 1 byte/elem
    const int*           gp = (const int*)d_in[4];            // group_size scalar
    float*               op = (float*)d_out;

    const int ntiles = Bc * Hc * (Nc / 16);           // 2048 query tiles
    const int blocks = (ntiles + WAVES - 1) / WAVES;  // 8 waves (256 thr) per block
    attn_gc_kernel<<<blocks, 32 * WAVES, 0, stream>>>(qp, kp, vp, mp, gp, op);
}
